// RiskGCN_35218731827640
// MI455X (gfx1250) — matrix operand order repaired and checked by reading, not compiled, not run on hardware
//
#include <hip/hip_runtime.h>

#define HID 128

typedef __attribute__((ext_vector_type(16))) __bf16 v16bf;
typedef __attribute__((ext_vector_type(8)))  float  v8f;
typedef __attribute__((ext_vector_type(4)))  unsigned int v4u;
typedef __attribute__((ext_vector_type(8)))  unsigned int v8u;

union AFrag { v16bf v; unsigned short u[16]; };
union BFrag { v16bf v; uint4 q[2]; };

__device__ __forceinline__ unsigned short f2bf(float f) {
    union { float f; unsigned u; } c; c.f = f;
    unsigned u = c.u;
    u += 0x7fffu + ((u >> 16) & 1u);      // round-to-nearest-even
    return (unsigned short)(u >> 16);
}

// ---------------------------------------------------------------- utilities
__global__ void k_fill(float* __restrict__ p, float v, int n) {
    int i = blockIdx.x * blockDim.x + threadIdx.x;
    if (i < n) p[i] = v;
}

__global__ void k_deg_count(const int* __restrict__ dst, float* __restrict__ deg, int E) {
    int e = blockIdx.x * blockDim.x + threadIdx.x;
    if (e < E) unsafeAtomicAdd(&deg[dst[e]], 1.0f);
}

__global__ void k_dinv(const float* __restrict__ deg, float* __restrict__ dinv, int n) {
    int i = blockIdx.x * blockDim.x + threadIdx.x;
    if (i < n) {
        float d = deg[i];
        dinv[i] = (d > 0.0f) ? rsqrtf(d) : 0.0f;
    }
}

// Convert W [128x128 f32, row-major K x N] into per-fragment bf16 B layout.
// Fragment index: (ntile, kchunk, lane) -> 16 contiguous bf16.
// B 32x16 layout, contiguous k order (as A): lanes 0-15 hold K=0..15 of the chunk (2 K per
// VGPR), lanes 16-31 hold K=16..31; column N = lane&15.
__global__ void k_convW(const float* __restrict__ W, unsigned short* __restrict__ Wbf) {
    int t = blockIdx.x * blockDim.x + threadIdx.x;   // 0..1023
    if (t >= 8 * 4 * 32) return;
    int lane = t & 31;
    int kc   = (t >> 5) & 3;
    int nt   = t >> 7;
    int n    = nt * 16 + (lane & 15);
    int hi   = lane >> 4;
    unsigned short* o = Wbf + t * 16;
#pragma unroll
    for (int j = 0; j < 16; ++j) {
        int k = kc * 32 + hi * 16 + j;
        o[j] = f2bf(W[k * HID + n]);
    }
}

// ---------------------------------------------------------------- WMMA GEMM
// T[M x 128] = A[M x 128] (f32 -> bf16 in-flight) @ Wbf (pre-laid fragments).
// Per block: TDM stages the whole 32KB B-fragment buffer into LDS once, then
// 8 waves each compute a 16-row x 128-col tile: 4 A-fragments reused across 8
// N-tiles, 32 x v_wmma_f32_16x16x32_bf16 per wave, f32 accumulate.
__global__ void __launch_bounds__(256)
k_gemm_wmma(const float* __restrict__ A, const unsigned short* __restrict__ Wbf,
            float* __restrict__ T, int M)
{
    __shared__ uint4 sW[2048];             // 32 KB staged B fragments

    const int lane = threadIdx.x & 31;
    const int wv   = threadIdx.x >> 5;

    // --- Tensor Data Mover: wave 0 DMAs 32KB global -> LDS (one 2D D#) ---
    if (wv == 0) {
        void* lp = (void*)sW;              // generic ptr; low 32 bits = LDS offset
        unsigned long long ga = (unsigned long long)(uintptr_t)Wbf;
        unsigned ldsoff = (unsigned)(uintptr_t)lp;
        v4u g0; v8u g1;
        g0[0] = 1u;                                   // count=1, user D#
        g0[1] = ldsoff;                               // lds_addr (bytes)
        g0[2] = (unsigned)ga;                         // global_addr[31:0]
        g0[3] = (unsigned)((ga >> 32) & 0x1FFFFFFu)   // global_addr[56:32]
              | (2u << 30);                           // type = 2 (image)
        g1[0] = 0x10000u;      // wg_mask=0, data_size=1 (2 bytes)
        g1[1] = 0x40000000u;   // tensor_dim0 = 16384 (bits 63:48)
        g1[2] = 0x00010000u;   // tensor_dim1 = 1     (bits 95:80)
        g1[3] = 0x40000000u;   // tile_dim0   = 16384 (bits 127:112)
        g1[4] = 0x00000001u;   // tile_dim1 = 1, tile_dim2 = 0
        g1[5] = 16384u;        // tensor_dim0_stride[31:0]
        g1[6] = 0x40000000u;   // t0_stride[47:32]=0, t1_stride[15:0]=16384
        g1[7] = 0u;            // t1_stride[47:16]
        // lp passed as an operand so the LDS object escapes: the "memory"
        // clobber then legally covers the TDM's writes into sW.
        asm volatile("tensor_load_to_lds %0, %1"
                     :: "s"(g0), "s"(g1), "s"(lp) : "memory");
        __builtin_amdgcn_s_wait_tensorcnt(0);
    }
    __syncthreads();

    const int tile  = blockIdx.x * 8 + wv;
    const int mbase = tile * 16;
    if (mbase < M) {                       // wave-uniform: EXEC stays all-1s
        const int m16 = lane & 15;
        const int hi  = lane >> 4;
        int row = mbase + m16;
        if (row >= M) row = M - 1;         // clamp loads; stores guarded below
        const float* arow = A + row * HID;

        // A 16x32 layout (contiguous, matches B): lane holds K = hi*16 .. hi*16+15
        v16bf afr[4];
#pragma unroll
        for (int kc = 0; kc < 4; ++kc) {
            AFrag fr;
            const float4* p0 = reinterpret_cast<const float4*>(arow + kc * 32 + hi * 16);
            const float4* p1 = reinterpret_cast<const float4*>(arow + kc * 32 + hi * 16 + 8);
            float4 a0 = p0[0], a1 = p0[1], a2 = p1[0], a3 = p1[1];
            fr.u[0]  = f2bf(a0.x); fr.u[1]  = f2bf(a0.y); fr.u[2]  = f2bf(a0.z); fr.u[3]  = f2bf(a0.w);
            fr.u[4]  = f2bf(a1.x); fr.u[5]  = f2bf(a1.y); fr.u[6]  = f2bf(a1.z); fr.u[7]  = f2bf(a1.w);
            fr.u[8]  = f2bf(a2.x); fr.u[9]  = f2bf(a2.y); fr.u[10] = f2bf(a2.z); fr.u[11] = f2bf(a2.w);
            fr.u[12] = f2bf(a3.x); fr.u[13] = f2bf(a3.y); fr.u[14] = f2bf(a3.z); fr.u[15] = f2bf(a3.w);
            afr[kc] = fr.v;
        }

        const bool full = (mbase + 16 <= M);   // wave-uniform
#pragma unroll
        for (int nt = 0; nt < 8; ++nt) {
            v8f c = {};
#pragma unroll
            for (int kc = 0; kc < 4; ++kc) {
                BFrag fb;
                const int frag = ((nt * 4 + kc) * 32 + lane) * 2;
                fb.q[0] = sW[frag];
                fb.q[1] = sW[frag + 1];
                c = __builtin_amdgcn_wmma_f32_16x16x32_bf16(
                        false, afr[kc], false, fb.v, (short)0, c, false, false);
            }
            // C/D layout: N = lane&15, VGPR r -> M = r + (lane>>4)*8
            float* tp = T + (mbase + hi * 8) * HID + nt * 16 + m16;
            if (full) {
#pragma unroll
                for (int r = 0; r < 8; ++r) tp[r * HID] = c[r];
            } else {
#pragma unroll
                for (int r = 0; r < 8; ++r)
                    if (mbase + hi * 8 + r < M) tp[r * HID] = c[r];
            }
        }
    }
}

// ---------------------------------------------------------------- aggregation
// agg[i] = t[i] * dinv[node]^2   (self-loop term doubles as initializer)
__global__ void k_self_init(const float4* __restrict__ T4, const float* __restrict__ dinv,
                            float4* __restrict__ agg4, int total4) {
    int i = blockIdx.x * blockDim.x + threadIdx.x;
    if (i < total4) {
        float di = dinv[i >> 5];
        float w = di * di;
        float4 t = T4[i];
        t.x *= w; t.y *= w; t.z *= w; t.w *= w;
        agg4[i] = t;
    }
}

// One wave per edge: gather 512B of h[src], scale by dinv[s]*dinv[d],
// HW fp32 atomic scatter-add into agg[dst].
__global__ void k_edge_scatter(const int* __restrict__ src, const int* __restrict__ dst,
                               const float* __restrict__ dinv,
                               const float* __restrict__ T, float* __restrict__ agg, int E) {
    int gid  = blockIdx.x * blockDim.x + threadIdx.x;
    int lane = gid & 31;
    int e    = gid >> 5;                   // wave-uniform
    if (e >= E) return;
    int s = src[e], d = dst[e];
    float w = dinv[s] * dinv[d];
    float4 v = reinterpret_cast<const float4*>(T + s * HID)[lane];
    float* o = agg + d * HID + lane * 4;
    unsafeAtomicAdd(o + 0, v.x * w);
    unsafeAtomicAdd(o + 1, v.y * w);
    unsafeAtomicAdd(o + 2, v.z * w);
    unsafeAtomicAdd(o + 3, v.w * w);
}

__global__ void k_bias_relu(const float4* __restrict__ agg4, const float4* __restrict__ b4,
                            float4* __restrict__ h4, int total4) {
    int i = blockIdx.x * blockDim.x + threadIdx.x;
    if (i < total4) {
        float4 v = agg4[i];
        float4 b = b4[i & 31];
        v.x = fmaxf(v.x + b.x, 0.0f);
        v.y = fmaxf(v.y + b.y, 0.0f);
        v.z = fmaxf(v.z + b.z, 0.0f);
        v.w = fmaxf(v.w + b.w, 0.0f);
        h4[i] = v;
    }
}

// ---------------------------------------------------------------- pooling
__global__ void k_pool_cnt(const int* __restrict__ batch, float* __restrict__ gcnt, int n) {
    int i = blockIdx.x * blockDim.x + threadIdx.x;
    if (i < n) unsafeAtomicAdd(&gcnt[batch[i]], 1.0f);
}

__global__ void k_pool_accum(const float4* __restrict__ h4, const int* __restrict__ batch,
                             float* __restrict__ gsum, int total4) {
    int i = blockIdx.x * blockDim.x + threadIdx.x;
    if (i < total4) {
        int node = i >> 5;
        int f = (i & 31) * 4;
        float4 v = h4[i];
        float* o = gsum + batch[node] * HID + f;
        unsafeAtomicAdd(o + 0, v.x);
        unsafeAtomicAdd(o + 1, v.y);
        unsafeAtomicAdd(o + 2, v.z);
        unsafeAtomicAdd(o + 3, v.w);
    }
}

__global__ void k_head(const float* __restrict__ gsum, const float* __restrict__ gcnt,
                       const float* __restrict__ Wh, const float* __restrict__ bh,
                       float* __restrict__ out) {
    int g = threadIdx.x;                   // 256 graphs, one block
    float inv = 1.0f / fmaxf(gcnt[g], 1.0f);
    float a0 = 0.0f, a1 = 0.0f;
    for (int f = 0; f < HID; ++f) {
        float p = gsum[g * HID + f] * inv;
        a0 += p * Wh[f * 2 + 0];
        a1 += p * Wh[f * 2 + 1];
    }
    out[g * 2 + 0] = a0 + bh[0];
    out[g * 2 + 1] = a1 + bh[1];
}

// ---------------------------------------------------------------- launcher
extern "C" void kernel_launch(void* const* d_in, const int* in_sizes, int n_in,
                              void* d_out, int out_size, void* d_ws, size_t ws_size,
                              hipStream_t stream) {
    (void)n_in; (void)out_size; (void)ws_size;
    const float* x   = (const float*)d_in[0];
    const int*   ei  = (const int*)d_in[1];
    const int*   bat = (const int*)d_in[2];
    const float* W1  = (const float*)d_in[3];
    const float* b1  = (const float*)d_in[4];
    const float* W2  = (const float*)d_in[5];
    const float* b2  = (const float*)d_in[6];
    const float* W3  = (const float*)d_in[7];
    const float* b3  = (const float*)d_in[8];
    const float* Wh  = (const float*)d_in[9];
    const float* bh  = (const float*)d_in[10];
    float* out = (float*)d_out;

    const int N = in_sizes[0] / HID;
    const int E = in_sizes[1] / 2;
    const int* src = ei;
    const int* dst = ei + E;

    // workspace carve-up (256B aligned)
    char* ws = (char*)d_ws;
    size_t off = 0;
    auto carve = [&](size_t bytes) -> void* {
        void* p = ws + off;
        off += (bytes + 255) & ~(size_t)255;
        return p;
    };
    const int total  = N * HID;                      // 12.8M, fits int
    const int total4 = total / 4;
    float*          buf0 = (float*)carve((size_t)total * 4);
    float*          buf1 = (float*)carve((size_t)total * 4);
    float*          deg  = (float*)carve((size_t)N * 4);
    float*          dinv = (float*)carve((size_t)N * 4);
    unsigned short* Wbf  = (unsigned short*)carve(8 * 4 * 32 * 16 * 2);
    float*          gsum = (float*)carve(256 * HID * 4);
    float*          gcnt = (float*)carve(256 * 4);

    const int TB = 256;
    const int gN  = (N + TB - 1) / TB;
    const int gE  = (E + TB - 1) / TB;
    const int gF4 = (total4 + TB - 1) / TB;
    const int gEw = (E + 7) / 8;                     // 8 edge-waves per block
    const int tiles = (N + 15) / 16;
    const int gG  = (tiles + 7) / 8;                 // 8 wave-tiles per block

    // degree / normalization (self-loop contributes the initial 1.0)
    k_fill<<<gN, TB, 0, stream>>>(deg, 1.0f, N);
    k_deg_count<<<gE, TB, 0, stream>>>(dst, deg, E);
    k_dinv<<<gN, TB, 0, stream>>>(deg, dinv, N);

    // layer 1: x -> buf0 -> buf1 -> buf0
    k_convW<<<4, TB, 0, stream>>>(W1, Wbf);
    k_gemm_wmma<<<gG, TB, 0, stream>>>(x, Wbf, buf0, N);
    k_self_init<<<gF4, TB, 0, stream>>>((const float4*)buf0, dinv, (float4*)buf1, total4);
    k_edge_scatter<<<gEw, TB, 0, stream>>>(src, dst, dinv, buf0, buf1, E);
    k_bias_relu<<<gF4, TB, 0, stream>>>((const float4*)buf1, (const float4*)b1, (float4*)buf0, total4);

    // layer 2: buf0 -> buf1 -> buf0 -> buf1
    k_convW<<<4, TB, 0, stream>>>(W2, Wbf);
    k_gemm_wmma<<<gG, TB, 0, stream>>>(buf0, Wbf, buf1, N);
    k_self_init<<<gF4, TB, 0, stream>>>((const float4*)buf1, dinv, (float4*)buf0, total4);
    k_edge_scatter<<<gEw, TB, 0, stream>>>(src, dst, dinv, buf1, buf0, E);
    k_bias_relu<<<gF4, TB, 0, stream>>>((const float4*)buf0, (const float4*)b2, (float4*)buf1, total4);

    // layer 3: buf1 -> buf0 -> buf1 -> buf0
    k_convW<<<4, TB, 0, stream>>>(W3, Wbf);
    k_gemm_wmma<<<gG, TB, 0, stream>>>(buf1, Wbf, buf0, N);
    k_self_init<<<gF4, TB, 0, stream>>>((const float4*)buf0, dinv, (float4*)buf1, total4);
    k_edge_scatter<<<gEw, TB, 0, stream>>>(src, dst, dinv, buf0, buf1, E);
    k_bias_relu<<<gF4, TB, 0, stream>>>((const float4*)buf1, (const float4*)b3, (float4*)buf0, total4);

    // global mean pool + head
    k_fill<<<(256 * HID + TB - 1) / TB, TB, 0, stream>>>(gsum, 0.0f, 256 * HID);
    k_fill<<<1, TB, 0, stream>>>(gcnt, 0.0f, 256);
    k_pool_cnt<<<gN, TB, 0, stream>>>(bat, gcnt, N);
    k_pool_accum<<<gF4, TB, 0, stream>>>((const float4*)buf0, bat, gsum, total4);
    k_head<<<1, 256, 0, stream>>>(gsum, gcnt, Wh, bh, out);
}
